// FineGrainedCellLSTM_38414187495770
// MI455X (gfx1250) — matrix operand order repaired and checked by reading, not compiled, not run on hardware
//
#include <hip/hip_runtime.h>
#include <hip/hip_bf16.h>
#include <stdint.h>

// ---------------------------------------------------------------------------
// LSTM forward for MI455X (gfx1250, wave32, WMMA).
//   T=512, B=64, I=512, H=512.
// Phase 1: xg[T*B,4H] = bf16(x) @ Wx + (bx+bh)        (v_wmma_f32_16x16x32_bf16)
// Phase 2: persistent kernel, 512 steps, 16 workgroups, software grid barrier.
// ---------------------------------------------------------------------------

#define Tn 512
#define Bn 64
#define In 512
#define Hn 512
#define G4 2048          // 4*H
#define NWG 16           // workgroups in recurrent kernel

typedef __attribute__((ext_vector_type(16))) __bf16 v16bf;
typedef __attribute__((ext_vector_type(8)))  float  v8f;
typedef unsigned short bf16_t;

union Frag { v16bf v; uint4 u[2]; };

__device__ __forceinline__ bf16_t f2bf(float f) {
  // round-to-nearest-even f32 -> bf16
  uint32_t u = __float_as_uint(f);
  uint32_t r = 0x7FFFu + ((u >> 16) & 1u);
  return (bf16_t)((u + r) >> 16);
}

__device__ __forceinline__ float sigmoidf_(float x) {
  return 1.0f / (1.0f + __expf(-x));
}

// ------------------------- workspace layout (bytes) -------------------------
#define OFF_XG    ((size_t)0)                               // f32 [T*B,2048]  256 MB
#define OFF_XBF   (OFF_XG   + (size_t)Tn*Bn*G4*4)           // bf16 [T*B,512]   32 MB
#define OFF_WXT   (OFF_XBF  + (size_t)Tn*Bn*In*2)           // bf16 [2048,512]   2 MB
#define OFF_WHT   (OFF_WXT  + (size_t)G4*In*2)              // bf16 [2048,512]   2 MB
#define OFF_BIAS  (OFF_WHT  + (size_t)G4*Hn*2)              // f32  [2048]
#define OFF_H0B   (OFF_BIAS + (size_t)G4*4)                 // bf16 [64,512] ping
#define OFF_H1B   (OFF_H0B  + (size_t)Bn*Hn*2)              // bf16 [64,512] pong
#define OFF_C     (OFF_H1B  + (size_t)Bn*Hn*2)              // f32  [64,512]
#define OFF_BAR   (OFF_C    + (size_t)Bn*Hn*4)              // barrier ints
#define WS_NEEDED (OFF_BAR + 256)

// --------------------------- prep: weights + bias ---------------------------
__global__ __launch_bounds__(256) void lstm_prep_weights(
    const float* __restrict__ Wxi, const float* __restrict__ Wxf,
    const float* __restrict__ Wxg, const float* __restrict__ Wxo,
    const float* __restrict__ Whi, const float* __restrict__ Whf,
    const float* __restrict__ Whg, const float* __restrict__ Who,
    const float* __restrict__ bxi, const float* __restrict__ bxf,
    const float* __restrict__ bxg, const float* __restrict__ bxo,
    const float* __restrict__ bhi, const float* __restrict__ bhf,
    const float* __restrict__ bhg, const float* __restrict__ bho,
    bf16_t* __restrict__ WxT, bf16_t* __restrict__ WhT, float* __restrict__ bias)
{
  int idx = blockIdx.x * 256 + threadIdx.x;       // 0 .. 2048*512-1
  int col = idx >> 9;                              // transposed row: gate col
  int k   = idx & 511;                             // K index
  int g = col >> 9, j = col & 511;
  const float* Wx = (g == 0) ? Wxi : (g == 1) ? Wxf : (g == 2) ? Wxg : Wxo;
  const float* Wh = (g == 0) ? Whi : (g == 1) ? Whf : (g == 2) ? Whg : Who;
  WxT[idx] = f2bf(Wx[(size_t)k * Hn + j]);
  WhT[idx] = f2bf(Wh[(size_t)k * Hn + j]);
  if (k == 0) {
    const float* bx = (g == 0) ? bxi : (g == 1) ? bxf : (g == 2) ? bxg : bxo;
    const float* bh = (g == 0) ? bhi : (g == 1) ? bhf : (g == 2) ? bhg : bho;
    bias[col] = bx[j] + bh[j];
  }
}

// ------------------------------ prep: x -> bf16 -----------------------------
__global__ __launch_bounds__(256) void lstm_prep_x(
    const float4* __restrict__ x, ushort4* __restrict__ xb)
{
  size_t i = (size_t)blockIdx.x * 256 + threadIdx.x;   // one float4 each
  float4 v = x[i];
  ushort4 o;
  o.x = f2bf(v.x); o.y = f2bf(v.y); o.z = f2bf(v.z); o.w = f2bf(v.w);
  xb[i] = o;
}

// ------------------------------ prep: h0 / c0 -------------------------------
__global__ __launch_bounds__(256) void lstm_prep_state(
    const float* __restrict__ h0, const float* __restrict__ c0,
    bf16_t* __restrict__ hbf0, float* __restrict__ cws)
{
  int i = blockIdx.x * 256 + threadIdx.x;              // 0 .. 64*512-1
  hbf0[i] = f2bf(h0[i]);
  cws[i]  = c0[i];
}

// --------------------- phase 1: xg = x @ Wx + bias (WMMA) -------------------
// grid (512, 16): block tile 64 rows x 128 cols, 8 waves = 4(m) x 2(n),
// each wave: 1 M-tile x 4 N-tiles, K = 512 in 16 chunks of 32.
__global__ __launch_bounds__(256) void lstm_gemm_xg(
    const bf16_t* __restrict__ xb,    // [T*B, 512]
    const bf16_t* __restrict__ WxT,   // [2048, 512]
    const float*  __restrict__ bias,  // [2048]
    float* __restrict__ xg)           // [T*B, 2048]
{
  const int lane = threadIdx.x & 31;
  const int wave = threadIdx.x >> 5;        // 0..7
  const int mw = wave & 3, nw = wave >> 2;  // 4 x 2
  const int m0 = blockIdx.x * 64 + mw * 16;
  const int n0 = blockIdx.y * 128 + nw * 64;
  const int l15 = lane & 15;
  const int hi  = lane >> 4;                // half-wave id

  v8f acc[4] = {v8f{}, v8f{}, v8f{}, v8f{}};
  const bf16_t* arow  = xb  + (size_t)(m0 + l15) * In;
  const bf16_t* brow0 = WxT + (size_t)(n0 + l15) * In;

  for (int kc = 0; kc < 16; ++kc) {
    const int kb = kc * 32;
    Frag a;
    // A layout: half-wave 0: K 0-15; half-wave 1: K 16-31 (contiguous)
    a.u[0] = *(const uint4*)(arow + kb + hi * 16);
    a.u[1] = *(const uint4*)(arow + kb + hi * 16 + 8);
    __builtin_prefetch(arow + kb + 64, 0, 0);   // global_prefetch next chunks
#pragma unroll
    for (int nt = 0; nt < 4; ++nt) {
      Frag b;
      // B layout: half-wave 0: K 0-15; half-wave 1: K 16-31 (contiguous)
      const bf16_t* bp = brow0 + (size_t)nt * 16 * In + kb + hi * 16;
      b.u[0] = ((const uint4*)bp)[0];
      b.u[1] = ((const uint4*)bp)[1];
      acc[nt] = __builtin_amdgcn_wmma_f32_16x16x32_bf16(
          false, a.v, false, b.v, (short)0, acc[nt], false, false);
    }
  }
#pragma unroll
  for (int nt = 0; nt < 4; ++nt) {
    const int col = n0 + nt * 16 + l15;
    const float bb = bias[col];
    const int rb = m0 + hi * 8;               // C layout: M = r + 8*half
#pragma unroll
    for (int r = 0; r < 8; ++r)
      xg[(size_t)(rb + r) * G4 + col] = acc[nt][r] + bb;
  }
}

// ------------- phase 2: 512-step recurrence, persistent kernel --------------
// 16 workgroups x 256 threads. WG owns 32 j-columns; wave (jt,mt) owns one
// (16 batch x 16 j) tile and computes ALL FOUR gate tiles for it, so the
// cell update is register-local. h is double-buffered across steps.
__global__ __launch_bounds__(256) void lstm_recurrent(
    const bf16_t* __restrict__ WhT,   // [2048, 512]
    const float*  __restrict__ xg,    // [T*B, 2048], bias folded in
    bf16_t* __restrict__ hbuf0, bf16_t* __restrict__ hbuf1,   // [64,512] each
    float* __restrict__ cws,          // [64,512]
    float* __restrict__ out,          // [T,64,512]
    float* __restrict__ hT, float* __restrict__ cT,           // [64,512]
    int* __restrict__ bar)            // bar[0]=count, bar[1]=generation
{
  const int lane = threadIdx.x & 31;
  const int wave = threadIdx.x >> 5;        // 0..7
  const int jt = wave & 1, mt = wave >> 1;  // 2 j-tiles x 4 m-tiles
  const int m0 = mt * 16;
  const int jbase = blockIdx.x * 32 + jt * 16;
  const int l15 = lane & 15;
  const int hi  = lane >> 4;
  const int jcol = jbase + l15;

  const bf16_t* brow[4];
#pragma unroll
  for (int g = 0; g < 4; ++g)
    brow[g] = WhT + (size_t)(g * Hn + jbase + l15) * Hn;

  for (int t = 0; t < Tn; ++t) {
    const bf16_t* hread  = (t & 1) ? hbuf1 : hbuf0;
    bf16_t*       hwrite = (t & 1) ? hbuf0 : hbuf1;

    v8f acc[4] = {v8f{}, v8f{}, v8f{}, v8f{}};
    const bf16_t* arow = hread + (size_t)(m0 + l15) * Hn;
    for (int kc = 0; kc < 16; ++kc) {
      const int kb = kc * 32;
      Frag a;
      a.u[0] = *(const uint4*)(arow + kb + hi * 16);
      a.u[1] = *(const uint4*)(arow + kb + hi * 16 + 8);
#pragma unroll
      for (int g = 0; g < 4; ++g) {
        Frag b;
        const bf16_t* bp = brow[g] + kb + hi * 16;
        b.u[0] = ((const uint4*)bp)[0];
        b.u[1] = ((const uint4*)bp)[1];
        acc[g] = __builtin_amdgcn_wmma_f32_16x16x32_bf16(
            false, a.v, false, b.v, (short)0, acc[g], false, false);
      }
    }

    // ------- elementwise cell update (register-local per lane) -------
    const float* xgt = xg + (size_t)t * Bn * G4;
#pragma unroll
    for (int r = 0; r < 8; ++r) {
      const int bi = m0 + hi * 8 + r;                 // batch row
      const size_t gx = (size_t)bi * G4 + jcol;
      const float gi = acc[0][r] + xgt[gx];
      const float gf = acc[1][r] + xgt[gx + 512];
      const float gg = acc[2][r] + xgt[gx + 1024];
      const float go = acc[3][r] + xgt[gx + 1536];
      const float i_ = sigmoidf_(gi);
      const float f_ = sigmoidf_(gf);
      const float g_ = tanhf(gg);
      const float o_ = sigmoidf_(go);
      const size_t hx = (size_t)bi * Hn + jcol;
      const float c_new = f_ * cws[hx] + i_ * g_;
      const float h_new = o_ * c_new;                 // NOTE: no tanh(c) per ref
      cws[hx] = c_new;
      out[(size_t)t * Bn * Hn + hx] = h_new;
      hwrite[hx] = f2bf(h_new);
      if (t == Tn - 1) { hT[hx] = h_new; cT[hx] = c_new; }
    }

    // ----------------------- grid-wide barrier -----------------------
    __builtin_amdgcn_fence(__ATOMIC_RELEASE, "agent");   // publish h/c writes
    __syncthreads();
    if (threadIdx.x == 0) {
      int ticket = __hip_atomic_fetch_add(&bar[0], 1, __ATOMIC_ACQ_REL,
                                          __HIP_MEMORY_SCOPE_AGENT);
      if (ticket == NWG - 1) {
        __hip_atomic_store(&bar[0], 0, __ATOMIC_RELAXED,
                           __HIP_MEMORY_SCOPE_AGENT);
        __hip_atomic_fetch_add(&bar[1], 1, __ATOMIC_RELEASE,
                               __HIP_MEMORY_SCOPE_AGENT);
      } else {
        while (__hip_atomic_load(&bar[1], __ATOMIC_ACQUIRE,
                                 __HIP_MEMORY_SCOPE_AGENT) <= t)
          __builtin_amdgcn_s_sleep(1);
      }
    }
    __syncthreads();
    __builtin_amdgcn_fence(__ATOMIC_ACQUIRE, "agent");   // see others' h writes
  }
}

// ------------------------------- host launch --------------------------------
extern "C" void kernel_launch(void* const* d_in, const int* in_sizes, int n_in,
                              void* d_out, int out_size, void* d_ws, size_t ws_size,
                              hipStream_t stream) {
  const float* x  = (const float*)d_in[0];
  const float* h0 = (const float*)d_in[1];
  const float* c0 = (const float*)d_in[2];
  // gate order i,f,g,o: W_x at 3,5,7,9 / b_x at 4,6,8,10 / W_h 11,13,15,17 / b_h 12,14,16,18
  const float* Wxi = (const float*)d_in[3];  const float* bxi = (const float*)d_in[4];
  const float* Wxf = (const float*)d_in[5];  const float* bxf = (const float*)d_in[6];
  const float* Wxg = (const float*)d_in[7];  const float* bxg = (const float*)d_in[8];
  const float* Wxo = (const float*)d_in[9];  const float* bxo = (const float*)d_in[10];
  const float* Whi = (const float*)d_in[11]; const float* bhi = (const float*)d_in[12];
  const float* Whf = (const float*)d_in[13]; const float* bhf = (const float*)d_in[14];
  const float* Whg = (const float*)d_in[15]; const float* bhg = (const float*)d_in[16];
  const float* Who = (const float*)d_in[17]; const float* bho = (const float*)d_in[18];

  char* ws = (char*)d_ws;
  float*  xg    = (float*) (ws + OFF_XG);
  bf16_t* xbf   = (bf16_t*)(ws + OFF_XBF);
  bf16_t* WxT   = (bf16_t*)(ws + OFF_WXT);
  bf16_t* WhT   = (bf16_t*)(ws + OFF_WHT);
  float*  bias  = (float*) (ws + OFF_BIAS);
  bf16_t* hbuf0 = (bf16_t*)(ws + OFF_H0B);
  bf16_t* hbuf1 = (bf16_t*)(ws + OFF_H1B);
  float*  cws   = (float*) (ws + OFF_C);
  int*    bar   = (int*)   (ws + OFF_BAR);

  float* out = (float*)d_out;                       // [T,B,H]
  float* hT  = out + (size_t)Tn * Bn * Hn;          // [B,H]
  float* cT  = hT + (size_t)Bn * Hn;                // [B,H]

  hipMemsetAsync(bar, 0, 256, stream);              // reset barrier every call

  lstm_prep_weights<<<(G4 * In) / 256, 256, 0, stream>>>(
      Wxi, Wxf, Wxg, Wxo, Whi, Whf, Whg, Who,
      bxi, bxf, bxg, bxo, bhi, bhf, bhg, bho, WxT, WhT, bias);

  lstm_prep_x<<<((size_t)Tn * Bn * In / 4) / 256, 256, 0, stream>>>(
      (const float4*)x, (ushort4*)xbf);

  lstm_prep_state<<<(Bn * Hn) / 256, 256, 0, stream>>>(h0, c0, hbuf0, cws);

  lstm_gemm_xg<<<dim3((Tn * Bn) / 64, G4 / 128), 256, 0, stream>>>(
      xbf, WxT, bias, xg);

  lstm_recurrent<<<NWG, 256, 0, stream>>>(
      WhT, xg, hbuf0, hbuf1, cws, out, hT, cT, bar);
}
